// LinearAttention_78683800863344
// MI455X (gfx1250) — matrix operand order repaired and checked by reading, not compiled, not run on hardware
//
#include <hip/hip_runtime.h>

// ---------------------------------------------------------------------------
// LinearAttention-style pipeline for MI455X (gfx1250, wave32, WMMA).
//   GEMM1: x0 = relu(W0[6144,1024] @ inp[1024,2048])     per batch
//   GEMM2: x1 = relu(sum_k W1k[2048,2048] @ shift(x0,k)) per batch,group (causal k=5)
//   GEMM3: x2 = W2[1024,2048] @ x1                       per batch,group -> fp32
//   EPI:   channel cumsum, /divisor*scale+shift, mean-center, L2-normalize
// Matrix math: v_wmma_f32_16x16x32_f16 (f32 accumulate).
// A staging: TDM tensor_load_to_lds (conv, 5 descriptors/chunk, HW-padded LDS
// pitch) or global_load_async_to_lds_b128 (plain GEMMs).
// Conv taps share one staged B tile per K-chunk via a 16-column LDS halo.
// ---------------------------------------------------------------------------

typedef __attribute__((ext_vector_type(16))) _Float16 v16h;
typedef __attribute__((ext_vector_type(8)))  float    v8f;
typedef __attribute__((ext_vector_type(4)))  unsigned u32x4;
typedef __attribute__((ext_vector_type(8)))  unsigned u32x8;

union F16Frag { v16h v; uint4 u[2]; };

#define BM 128
#define BN 128
#define BK 32
#define LDP 40   // padded LDS row pitch in halves (80B rows -> conflict-free b128)

// ---------------------------------------------------------------------------
// TDM 2D tile load: global (row-major, f16) -> LDS, optional LDS row padding.
// D# per CDNA5 ISA ch.8: group0 = {count/flags, lds_addr, global_addr, type=2},
// group1 = {data_size/pad cfg, tensor dims, tile dims, dim0 stride}.
// 2-group form => 2D tensor (VADDR2/3 null).
// ---------------------------------------------------------------------------
__device__ __forceinline__ void tdm_load_2d_f16(
    unsigned lds_byte_addr, const void* gaddr,
    unsigned tile_d0_elems, unsigned tile_d1_rows,
    unsigned long long row_stride_elems,
    unsigned pad_interval_code, unsigned pad_amount_code, bool pad_en) {
  const unsigned long long ga = (unsigned long long)gaddr;
  u32x4 g0;
  g0.x = 1u;                                    // count=1 (valid user D#)
  g0.y = lds_byte_addr;                         // [63:32] lds_addr
  g0.z = (unsigned)ga;                          // [95:64] global_addr lo
  g0.w = (unsigned)(ga >> 32) | 0x80000000u;    // [120:96] addr hi, [127:126]=2
  u32x8 g1;
  g1.s0 = (1u << 16)                                    // data_size = 2B
        | (pad_en ? (1u << 20) : 0u)                    // pad_enable
        | (pad_interval_code << 22)                     // pad every 2^(c+1) dwords
        | (pad_amount_code << 25);                      // pad (c+1) dwords
  g1.s1 = (tile_d0_elems & 0xFFFFu) << 16;              // tensor_dim0 = tile_d0
  g1.s2 = (tile_d0_elems >> 16) | ((tile_d1_rows & 0xFFFFu) << 16); // tensor_dim1
  g1.s3 = (tile_d1_rows >> 16) | (tile_d0_elems << 16); // tile_dim0
  g1.s4 = tile_d1_rows;                                 // tile_dim1 (tile_dim2=0)
  g1.s5 = (unsigned)row_stride_elems;                   // tensor_dim0_stride lo32
  g1.s6 = (unsigned)(row_stride_elems >> 32) & 0xFFFFu; // stride hi16
  g1.s7 = 0u;
  asm volatile("tensor_load_to_lds %0, %1" :: "s"(g0), "s"(g1) : "memory");
}

// ---------------------------------------------------------------------------
__global__ void f32_to_f16_kernel(const float* __restrict__ in,
                                  _Float16* __restrict__ out, size_t n) {
  size_t stride = (size_t)gridDim.x * blockDim.x;
  for (size_t i = (size_t)blockIdx.x * blockDim.x + threadIdx.x; i < n; i += stride)
    out[i] = (_Float16)in[i];
}

// w1: [6144][2048][5] fp32  ->  w1p: [5][6144][2048] fp16 (tap-major rows)
__global__ void repack_w1_kernel(const float* __restrict__ w1,
                                 _Float16* __restrict__ out, size_t total) {
  size_t stride = (size_t)gridDim.x * blockDim.x;
  for (size_t i = (size_t)blockIdx.x * blockDim.x + threadIdx.x; i < total; i += stride) {
    size_t k   = i % 5;
    size_t cin = (i / 5) % 2048;
    size_t m   = i / (5 * 2048);
    out[k * (size_t)6144 * 2048 + m * 2048 + cin] = (_Float16)w1[i];
  }
}

// ---------------------------------------------------------------------------
// Tiled WMMA GEMM, C[M,N] (+)= sum_tap A_tap[M,K] * shift(B[K,N], tap-NTAPS+1)
//  - block tile 128x128, 8 waves, wave tile 64x32 (4x2 x 16x16 WMMA subtiles)
//  - NTAPS=1: plain GEMM, A staged with per-lane async-to-LDS b128.
//  - NTAPS=5: causal conv; A staged by TDM (1 descriptor per tap, wave 0 only,
//    LDS pitch padding done by the TDM), B staged once per K-chunk with a
//    16-column left halo; taps read LDS rows at shifted offsets.
// ---------------------------------------------------------------------------
template <int NTAPS, bool RELU, bool OUTF16>
__global__ __launch_bounds__(256) void wmma_gemm_kernel(
    const _Float16* __restrict__ A, const _Float16* __restrict__ B,
    void* __restrict__ Cout,
    int N, int K, int groups,
    long aGroupStride, long aTapStride,
    long bBatchStride, long bGroupStride,
    long cBatchStride, long cGroupStride) {
  constexpr int BEXT  = (NTAPS > 1) ? 16 : 0;  // left halo columns for causal shifts
  constexpr int BROWS = BN + BEXT;             // transposed-B rows in LDS
  constexpr int NCH   = BROWS / 16;            // 16-column staging chunks per row

  __shared__ _Float16 As[NTAPS * BM * LDP];    // [tap*BM + m][k], 80B row pitch
  __shared__ _Float16 Bs[BROWS * LDP];         // [halo + n][k] (transposed)

  const int tid  = threadIdx.x;
  const int lane = tid & 31;
  const int wid  = tid >> 5;
  const int m_l  = lane & 15;
  const int hb   = lane >> 4;

  const int n0 = blockIdx.x * BN;
  const int m0 = blockIdx.y * BM;
  const int z  = blockIdx.z;
  const int b  = z / groups;
  const int g  = z % groups;

  const _Float16* Ag = A + (size_t)g * aGroupStride;
  const _Float16* Bg = B + (size_t)b * bBatchStride + (size_t)g * bGroupStride;
  const size_t cOff  = (size_t)b * cBatchStride + (size_t)g * cGroupStride;

  const int wm = (wid >> 2) * 64;  // wave M offset in block tile
  const int wn = (wid & 3) * 32;   // wave N offset in block tile

  // scalar (SGPR) wave id for uniform TDM-issuer branch (TDM ignores EXEC)
  const int swid = __builtin_amdgcn_readfirstlane(wid);

  v8f acc[4][2] = {};

  for (int k0 = 0; k0 < K; k0 += BK) {
    if constexpr (NTAPS > 1) {
      // ---- stage A tiles via Tensor Data Mover: one 2D descriptor per tap.
      // 128 rows x 64B, row stride K elems; pad 4 dwords every 16 dwords
      // reproduces the LDP=40 (80B) LDS pitch in hardware.
      if (swid == 0) {
        #pragma unroll
        for (int tap = 0; tap < NTAPS; ++tap) {
          tdm_load_2d_f16((unsigned)(size_t)&As[tap * BM * LDP],
                          Ag + (size_t)tap * aTapStride + (size_t)m0 * K + k0,
                          BK, BM, (unsigned long long)K,
                          /*pad_interval 16dw*/ 3u, /*pad 4dw*/ 3u, true);
        }
      }
    } else {
      // ---- stage A tile via per-lane async DMA to LDS: 16B per issue ----
      for (int u = tid; u < BM * 4; u += 256) {
        const int row  = u >> 2;
        const int colh = (u & 3) * 8;       // 8-half (16B) column chunk
        const unsigned long long ga = (unsigned long long)(
            Ag + (size_t)(m0 + row) * K + (k0 + colh));
        const unsigned lds = (unsigned)(size_t)&As[row * LDP + colh];
        asm volatile("global_load_async_to_lds_b128 %0, %1, off"
                     :: "v"(lds), "v"(ga) : "memory");
      }
    }
    // ---- stage B transposed with halo: Bs[BEXT+n][k] = B[k0+k][n0+n] ----
    for (int u = tid; u < BK * NCH; u += 256) {
      const int k    = u / NCH;
      const int c    = u % NCH;
      const int gcol = n0 - BEXT + c * 16;     // multiple of 16 -> 32B aligned
      _Float16 vals[16];
      if (gcol >= 0) {
        const uint4* s4 = (const uint4*)(Bg + (size_t)(k0 + k) * N + gcol);
        *(uint4*)&vals[0] = s4[0];
        *(uint4*)&vals[8] = s4[1];
      } else {                                  // fully-negative halo chunk
        #pragma unroll
        for (int i = 0; i < 16; ++i) vals[i] = (_Float16)0.0f;
      }
      #pragma unroll
      for (int i = 0; i < 16; ++i)
        Bs[(c * 16 + i) * LDP + k] = vals[i];
    }
    // ---- warm L2 with next K-chunk of B (weights live in the 192MB L2) ----
    if (k0 + BK < K)
      __builtin_prefetch(Bg + (size_t)(k0 + BK + (tid >> 3)) * N + n0 + (tid & 7) * 16,
                         0, 1);

    if constexpr (NTAPS > 1)
      __builtin_amdgcn_s_wait_tensorcnt(0);
    else
      asm volatile("s_wait_asynccnt 0x0" ::: "memory");
    __syncthreads();

    // ---- compute: all taps against the shared staged B ----
    #pragma unroll
    for (int tap = 0; tap < NTAPS; ++tap) {
      const int shift = tap - (NTAPS - 1);  // in [-(NTAPS-1), 0]
      F16Frag af[4], bf[2];
      #pragma unroll
      for (int i = 0; i < 4; ++i) {
        const int base = (tap * BM + wm + 16 * i + m_l) * LDP;
        af[i].u[0] = *(const uint4*)&As[base + 8 * hb];       // K 8*hb..+7
        af[i].u[1] = *(const uint4*)&As[base + 16 + 8 * hb];  // K 16+8*hb..+7
      }
      #pragma unroll
      for (int j = 0; j < 2; ++j) {
        const int base = (BEXT + wn + 16 * j + m_l + shift) * LDP + 8 * hb;
        bf[j].u[0] = *(const uint4*)&Bs[base];
        bf[j].u[1] = *(const uint4*)&Bs[base + 16];
      }
      #pragma unroll
      for (int i = 0; i < 4; ++i)
        #pragma unroll
        for (int j = 0; j < 2; ++j)
          acc[i][j] = __builtin_amdgcn_wmma_f32_16x16x32_f16(
              false, af[i].v, false, bf[j].v, (short)0, acc[i][j], false, false);
    }
    __syncthreads();
  }

  // ---- epilogue: relu + store (f16 or f32) ----
  float* C32 = (float*)Cout;
  _Float16* C16 = (_Float16*)Cout;
  #pragma unroll
  for (int i = 0; i < 4; ++i) {
    #pragma unroll
    for (int j = 0; j < 2; ++j) {
      #pragma unroll
      for (int r = 0; r < 8; ++r) {
        float v = acc[i][j][r];
        if (RELU) v = fmaxf(v, 0.0f);
        size_t row = (size_t)(m0 + wm + 16 * i + 8 * hb + r);
        size_t col = (size_t)(n0 + wn + 16 * j + m_l);
        size_t idx = cOff + row * (size_t)N + col;
        if (OUTF16) C16[idx] = (_Float16)v;
        else        C32[idx] = v;
      }
    }
  }
}

// ---------------------------------------------------------------------------
// Fused epilogue: one thread per (b,s) column, two streaming passes.
// Centered L2 norm derived from sum / sum-of-squares (no temp storage).
// Lane-adjacent s => fully coalesced channel-strided accesses.
// ---------------------------------------------------------------------------
__global__ __launch_bounds__(256) void epilogue_kernel(
    const float* __restrict__ x2, const float* __restrict__ divisor,
    float* __restrict__ out) {
  const int col = blockIdx.x * blockDim.x + threadIdx.x;  // 0..B*S-1
  const int b = col >> 11;           // S = 2048
  const int s = col & 2047;
  const size_t CS = 2048;            // channel stride in floats
  const float* dep = x2 + (size_t)b * 3072 * CS + s;
  const float* scl = dep + (size_t)1024 * CS;
  const float* shf = dep + (size_t)2048 * CS;
  const float invd = 1.0f / divisor[s];

  float run = 0.f, sum = 0.f, sumsq = 0.f;
  for (int c = 0; c < 1024; ++c) {
    float d = dep[(size_t)c * CS];
    run += d;
    float o = run * invd * scl[(size_t)c * CS] + shf[(size_t)c * CS];
    sum += o;
    sumsq += o * o;
  }
  const float mean   = sum * (1.0f / 1024.0f);
  const float varsum = fmaxf(sumsq - 1024.0f * mean * mean, 0.0f);
  const float denom  = sqrtf(varsum) * 0.03125f + 1e-5f;  // 1024^-0.5 = 1/32
  const float oscale = 0.25f / denom;                     // INIT_SCALE / denom

  run = 0.f;
  float* ob = out + (size_t)b * 1024 * CS + s;
  for (int c = 0; c < 1024; ++c) {
    float d = dep[(size_t)c * CS];
    run += d;
    float o = run * invd * scl[(size_t)c * CS] + shf[(size_t)c * CS];
    ob[(size_t)c * CS] = (o - mean) * oscale;
  }
}

// ---------------------------------------------------------------------------
extern "C" void kernel_launch(void* const* d_in, const int* in_sizes, int n_in,
                              void* d_out, int out_size, void* d_ws, size_t ws_size,
                              hipStream_t stream) {
  (void)in_sizes; (void)n_in; (void)out_size; (void)ws_size;
  const float* inp     = (const float*)d_in[0];
  const float* divisor = (const float*)d_in[1];
  const float* w0      = (const float*)d_in[2];
  const float* w1      = (const float*)d_in[3];
  const float* w2      = (const float*)d_in[4];
  float* out = (float*)d_out;

  const size_t B = 4, F = 1024, S = 2048, G = 3, K5 = 5;
  const size_t INTER = F * 2 * G;        // 6144
  const size_t CG = INTER / G;           // 2048 channels per group

  char* ws = (char*)d_ws;
  size_t off = 0;
  auto carve = [&](size_t bytes) -> void* {
    void* p = ws + off;
    off += (bytes + 255) & ~(size_t)255;
    return p;
  };
  _Float16* x_h  = (_Float16*)carve(B * F * S * 2);          //  16.8 MB
  _Float16* w0_h = (_Float16*)carve(INTER * F * 2);          //  12.6 MB
  _Float16* w1p  = (_Float16*)carve(K5 * INTER * CG * 2);    // 125.8 MB
  _Float16* w2_h = (_Float16*)carve(G * F * CG * 2);         //  12.6 MB
  _Float16* x0_h = (_Float16*)carve(B * INTER * S * 2);      // 100.7 MB
  _Float16* x1_h = (_Float16*)carve(B * INTER * S * 2);      // 100.7 MB
  float*    x2_f = (float*)   carve(B * G * F * S * 4);      // 100.7 MB

  // -- precision conversion / weight repack --
  f32_to_f16_kernel<<<4096, 256, 0, stream>>>(inp, x_h, B * F * S);
  f32_to_f16_kernel<<<2048, 256, 0, stream>>>(w0, w0_h, INTER * F);
  f32_to_f16_kernel<<<2048, 256, 0, stream>>>(w2, w2_h, G * F * CG);
  repack_w1_kernel<<<8192, 256, 0, stream>>>(w1, w1p, INTER * CG * K5);

  // -- GEMM1: x0 = relu(W0 @ inp), per batch --
  wmma_gemm_kernel<1, true, true><<<dim3(S / BN, INTER / BM, B), 256, 0, stream>>>(
      w0_h, x_h, x0_h,
      (int)S, (int)F, /*groups=*/1,
      0L, 0L,
      (long)(F * S), 0L,
      (long)(INTER * S), 0L);

  // -- GEMM2: causal grouped conv k=5 as 5 shifted GEMMs, per (batch,group) --
  wmma_gemm_kernel<5, true, true><<<dim3(S / BN, CG / BM, B * G), 256, 0, stream>>>(
      w1p, x0_h, x1_h,
      (int)S, (int)CG, /*groups=*/(int)G,
      (long)(CG * CG), (long)(INTER * CG),
      (long)(INTER * S), (long)(CG * S),
      (long)(INTER * S), (long)(CG * S));

  // -- GEMM3: x2 = W2 @ x1 (fp32 out for cumsum precision), per (batch,group) --
  wmma_gemm_kernel<1, false, false><<<dim3(S / BN, F / BM, B * G), 256, 0, stream>>>(
      w2_h, x1_h, x2_f,
      (int)S, (int)CG, /*groups=*/(int)G,
      (long)(F * CG), 0L,
      (long)(INTER * S), (long)(CG * S),
      (long)(G * F * S), (long)(F * S));

  // -- fused cumsum / scale / shift / center / L2-normalize --
  epilogue_kernel<<<(int)((B * S) / 256), 256, 0, stream>>>(x2_f, divisor, out);
}
